// LlamaSelfAttention_5781025981017
// MI455X (gfx1250) — matrix operand order repaired and checked by reading, not compiled, not run on hardware
//
#include <hip/hip_runtime.h>
#include <hip/hip_bf16.h>
#include <stdint.h>

typedef __attribute__((ext_vector_type(16))) __bf16 v16bf;
typedef __attribute__((ext_vector_type(8)))  float  v8f;

#define EMBED    4096
#define NHEADS   32
#define NKV      8
#define HEAD_DIM 128
#define QPERKV   4
#define B_       2
#define S_       2048
#define QKV_DIM  6144          /* (32 + 2*8) * 128 */
#define MROWS    (B_ * S_)     /* 4096 */
#define SCALE    0.08838834764831845f /* 1/sqrt(128) */

union AF { v16bf v; uint32_t u[8]; };

__device__ __forceinline__ uint16_t f2bf(float f) {
  uint32_t u = __float_as_uint(f);
  u += 0x7FFFu + ((u >> 16) & 1u);          // round-to-nearest-even
  return (uint16_t)(u >> 16);
}

// ---------------------------------------------------------------- fp32 -> bf16
__global__ void cvt_bf16_kernel(const float* __restrict__ src,
                                uint16_t* __restrict__ dst, int n) {
  int i = blockIdx.x * blockDim.x + threadIdx.x;
  if (i < n) dst[i] = f2bf(src[i]);
}

// ------------------------------------------------- C[M,N] = A[M,K] * B[N,K]^T
// A,B bf16 row-major (leading dim K); C fp32 row-major (leading dim N).
// One wave computes a 32(M) x 64(N) strip: 2 A-frags x 4 B-frags -> 8 WMMA
// per 12 b128 loads (1.5 loads/WMMA). Prefetch runs 2 K-blocks ahead.
__global__ __launch_bounds__(256)
void gemm_bt_wmma(const uint16_t* __restrict__ A, const uint16_t* __restrict__ B,
                  float* __restrict__ C, int M, int N, int K) {
  const int lane   = threadIdx.x & 31;
  const int wave   = (blockIdx.x * blockDim.x + threadIdx.x) >> 5;
  const int tilesN = N >> 6;
  const int tM     = wave / tilesN;       // 32-row strip
  const int tN     = wave % tilesN;       // 64-col strip
  if (tM * 32 >= M) return;               // wave-uniform

  const int halfSel = lane >> 4;          // 0 | 1
  const int l15     = lane & 15;
  const int kkA     = halfSel * 8;        // A: K {0-7,16-23} vs {8-15,24-31}
  const int kkB     = halfSel * 16;       // unused: B uses kkA order {0-7,16-23} vs {8-15,24-31}

  const uint32_t* Arow[2];
#pragma unroll
  for (int m = 0; m < 2; ++m)
    Arow[m] = (const uint32_t*)(A + (size_t)(tM * 32 + m * 16 + l15) * K);
  const uint32_t* Brow[4];
#pragma unroll
  for (int t = 0; t < 4; ++t)
    Brow[t] = (const uint32_t*)(B + (size_t)(tN * 64 + t * 16 + l15) * K);

  v8f acc[2][4] = {};
  for (int k0 = 0; k0 < K; k0 += 32) {
    if (k0 + 64 < K) {                    // warm WGP$/L2 two K-blocks ahead
      int kp = (k0 + 64) >> 1;
#pragma unroll
      for (int m = 0; m < 2; ++m) __builtin_prefetch(Arow[m] + kp, 0, 3);
#pragma unroll
      for (int t = 0; t < 4; ++t) __builtin_prefetch(Brow[t] + kp, 0, 3);
    }
    AF a[2], b[4];
#pragma unroll
    for (int m = 0; m < 2; ++m)
#pragma unroll
      for (int j = 0; j < 8; ++j) {
        int k = kkA + ((j < 4) ? (2 * j) : (16 + 2 * (j - 4)));
        a[m].u[j] = Arow[m][(k0 + k) >> 1];
      }
#pragma unroll
    for (int t = 0; t < 4; ++t)
#pragma unroll
      for (int j = 0; j < 8; ++j)
        b[t].u[j] = Brow[t][(k0 + kkA + ((j < 4) ? (2 * j) : (16 + 2 * (j - 4)))) >> 1];
#pragma unroll
    for (int m = 0; m < 2; ++m)
#pragma unroll
      for (int t = 0; t < 4; ++t)
        acc[m][t] = __builtin_amdgcn_wmma_f32_16x16x32_bf16(
            false, a[m].v, false, b[t].v, (short)0, acc[m][t], false, false);
  }
#pragma unroll
  for (int m = 0; m < 2; ++m)
#pragma unroll
    for (int t = 0; t < 4; ++t) {
      float* Cb = C + (size_t)(tM * 32 + m * 16 + halfSel * 8) * N
                    + tN * 64 + t * 16 + l15;
#pragma unroll
      for (int v = 0; v < 8; ++v)
        Cb[(size_t)v * N] = acc[m][t][v];
    }
}

// ---------------------------------------- RoPE on Q/K + scatter QKV head-major
// Q,K written head-major [b][h][s][d]; V written TRANSPOSED [b][g][d][s] so
// attention PV B-fragments load contiguous key-pairs.
__global__ void rope_scatter_kernel(const float* __restrict__ qkv,
                                    uint16_t* __restrict__ qb,
                                    uint16_t* __restrict__ kb,
                                    uint16_t* __restrict__ vb) {
  int tid  = blockIdx.x * blockDim.x + threadIdx.x;
  int pair = tid & 63;        tid >>= 6;
  int slot = tid % 6;         tid /= 6;   // 0..3 = q, 4 = k, 5 = v
  int g    = tid & 7;         tid >>= 3;
  int sPos = tid & (S_ - 1);  tid >>= 11;
  int bIdx = tid;
  if (bIdx >= B_) return;

  const float* src = qkv + (size_t)(bIdx * S_ + sPos) * QKV_DIM
                         + (g * 6 + slot) * HEAD_DIM + 2 * pair;
  float x1 = src[0], x2 = src[1];

  if (slot == 5) {  // V: transposed store [b][g][dim][s]
    uint16_t* dst = vb + ((size_t)(bIdx * NKV + g) * HEAD_DIM + 2 * pair) * S_ + sPos;
    dst[0]  = f2bf(x1);
    dst[S_] = f2bf(x2);
    return;
  }
  // inv_freq = 10000^(-2*pair/128)
  float inv_freq = expf(-(float)(2 * pair) * (9.210340371976184f / (float)HEAD_DIM));
  float ang = (float)sPos * inv_freq;
  float sn, c;
  sincosf(ang, &sn, &c);
  float o1 = x1 * c - x2 * sn;
  float o2 = x2 * c + x1 * sn;
  uint16_t* dst;
  if (slot == 4)
    dst = kb + ((size_t)(bIdx * NKV + g) * S_ + sPos) * HEAD_DIM + 2 * pair;
  else
    dst = qb + ((size_t)(bIdx * NHEADS + g * QPERKV + slot) * S_ + sPos) * HEAD_DIM + 2 * pair;
  dst[0] = f2bf(o1); dst[1] = f2bf(o2);
}

// ------------------------------------------------------- flash attention (bf16)
// One wave per (b, head, 16-row Q tile). Streams 32-key blocks: 8 WMMA for
// S=QK^T, online softmax, LDS round-trip to re-lay P as an A-fragment,
// 8 WMMA for O += P*V (V transposed -> vectorizable b128 fragment loads).
__global__ __launch_bounds__(256)
void flash_attn_kernel(const uint16_t* __restrict__ Q,
                       const uint16_t* __restrict__ Km,
                       const uint16_t* __restrict__ VmT,
                       uint16_t* __restrict__ O) {
  __shared__ float pbuf[8][16][33];
  const int lane = threadIdx.x & 31;
  const int w    = threadIdx.x >> 5;
  int id = blockIdx.x * 8 + w;
  const int qt   = id & 127;  id >>= 7;
  const int h    = id & 31;   id >>= 5;
  const int bIdx = id;
  if (bIdx >= B_) return;                    // grid is exact; wave-uniform
  const int g = h >> 2;

  const int l15     = lane & 15;
  const int halfSel = lane >> 4;
  const int kkA     = halfSel * 8;
  const int kkB     = halfSel * 16;

  const uint16_t* Qp  = Q   + ((size_t)(bIdx * NHEADS + h) * S_ + qt * 16) * HEAD_DIM;
  const uint16_t* Kp  = Km  +  (size_t)(bIdx * NKV   + g) * S_ * HEAD_DIM;
  const uint16_t* VpT = VmT +  (size_t)(bIdx * NKV   + g) * HEAD_DIM * S_;

  // Q fragments covering head_dim = 128 (4 x K=32)
  AF qa[4];
  {
    const uint32_t* Qrow = (const uint32_t*)(Qp + (size_t)l15 * HEAD_DIM);
#pragma unroll
    for (int s = 0; s < 4; ++s)
#pragma unroll
      for (int j = 0; j < 8; ++j) {
        int k = s * 32 + kkA + ((j < 4) ? (2 * j) : (16 + 2 * (j - 4)));
        qa[s].u[j] = Qrow[k >> 1];
      }
  }
  // Per-tile V row pointers (transposed layout: dim-major, keys contiguous)
  const uint32_t* Vrow[8];
#pragma unroll
  for (int t2 = 0; t2 < 8; ++t2)
    Vrow[t2] = (const uint32_t*)(VpT + (size_t)(t2 * 16 + l15) * S_);

  v8f acc[8] = {};
  float mrow[8], lrow[8];
#pragma unroll
  for (int v = 0; v < 8; ++v) { mrow[v] = -1e30f; lrow[v] = 0.f; }

  const int qrow = qt * 16 + halfSel * 8;    // + v = global row this lane holds
  const int nkb  = (qt * 16 + 15) / 32 + 1;  // causal bound on 32-key blocks

  for (int kbI = 0; kbI < nkb; ++kbI) {
    const int keyBase = kbI * 32;
    // ---- S = Q K^T (16 x 32)
    v8f s0 = {}, s1 = {};
    const uint32_t* Krow0 = (const uint32_t*)(Kp + (size_t)(keyBase + l15) * HEAD_DIM);
    const uint32_t* Krow1 = (const uint32_t*)(Kp + (size_t)(keyBase + 16 + l15) * HEAD_DIM);
#pragma unroll
    for (int s = 0; s < 4; ++s) {
      AF b0, b1;
#pragma unroll
      for (int j = 0; j < 8; ++j) {
        int k = s * 32 + kkA + ((j < 4) ? (2 * j) : (16 + 2 * (j - 4)));
        b0.u[j] = Krow0[k >> 1];
        b1.u[j] = Krow1[k >> 1];
      }
      s0 = __builtin_amdgcn_wmma_f32_16x16x32_bf16(false, qa[s].v, false, b0.v, (short)0, s0, false, false);
      s1 = __builtin_amdgcn_wmma_f32_16x16x32_bf16(false, qa[s].v, false, b1.v, (short)0, s1, false, false);
    }
    // ---- causal mask + scale + online softmax (rows live across 16-lane halves)
    const int key0 = keyBase + l15;
    const int key1 = keyBase + 16 + l15;
    float p0[8], p1[8];
#pragma unroll
    for (int v = 0; v < 8; ++v) {
      int r = qrow + v;
      float a0 = (key0 <= r) ? s0[v] * SCALE : -1e30f;
      float a1 = (key1 <= r) ? s1[v] * SCALE : -1e30f;
      float c = fmaxf(a0, a1);
      c = fmaxf(c, __shfl_xor(c, 1, 32));
      c = fmaxf(c, __shfl_xor(c, 2, 32));
      c = fmaxf(c, __shfl_xor(c, 4, 32));
      c = fmaxf(c, __shfl_xor(c, 8, 32));
      float mn    = fmaxf(mrow[v], c);
      float alpha = __expf(mrow[v] - mn);
      mrow[v] = mn;
      float e0 = __expf(a0 - mn);
      float e1 = __expf(a1 - mn);
      p0[v] = e0; p1[v] = e1;
      float rs = e0 + e1;
      rs += __shfl_xor(rs, 1, 32);
      rs += __shfl_xor(rs, 2, 32);
      rs += __shfl_xor(rs, 4, 32);
      rs += __shfl_xor(rs, 8, 32);
      lrow[v] = lrow[v] * alpha + rs;
#pragma unroll
      for (int t2 = 0; t2 < 8; ++t2) acc[t2][v] *= alpha;
    }
    // ---- stage P (C-layout) to LDS, read back in A-fragment layout as bf16
#pragma unroll
    for (int v = 0; v < 8; ++v) {
      int m = v + halfSel * 8;
      pbuf[w][m][l15]      = p0[v];
      pbuf[w][m][l15 + 16] = p1[v];
    }
    AF pf;
#pragma unroll
    for (int j = 0; j < 8; ++j) {
      int k = kkA + ((j < 4) ? (2 * j) : (16 + 2 * (j - 4)));
      uint32_t lo = f2bf(pbuf[w][l15][k]);
      uint32_t hi = f2bf(pbuf[w][l15][k + 1]);
      pf.u[j] = lo | (hi << 16);
    }
    // ---- O += P * V  (transposed V: contiguous key-pairs, b128-vectorizable)
#pragma unroll
    for (int t2 = 0; t2 < 8; ++t2) {
      AF vf;
#pragma unroll
      for (int j = 0; j < 8; ++j)
        vf.u[j] = Vrow[t2][(keyBase + kkA + ((j < 4) ? (2 * j) : (16 + 2 * (j - 4)))) >> 1];
      acc[t2] = __builtin_amdgcn_wmma_f32_16x16x32_bf16(false, pf.v, false, vf.v, (short)0, acc[t2], false, false);
    }
  }

  // ---- finalize: O /= l, write bf16 head-slice into [b,s,EMBED] layout
#pragma unroll
  for (int v = 0; v < 8; ++v) {
    float inv = 1.f / lrow[v];
    uint16_t* dst = O + (size_t)(bIdx * S_ + qrow + v) * EMBED + h * HEAD_DIM + l15;
#pragma unroll
    for (int t2 = 0; t2 < 8; ++t2)
      dst[t2 * 16] = f2bf(acc[t2][v] * inv);
  }
}

// ------------------------------------------------------------------- launcher
extern "C" void kernel_launch(void* const* d_in, const int* in_sizes, int n_in,
                              void* d_out, int out_size, void* d_ws, size_t ws_size,
                              hipStream_t stream) {
  const float* x     = (const float*)d_in[0];
  const float* w_qkv = (const float*)d_in[1];
  const float* w_out = (const float*)d_in[2];
  float* out = (float*)d_out;

  char* ws = (char*)d_ws;
  uint16_t* xb    = (uint16_t*)ws;  ws += (size_t)MROWS * EMBED * 2;
  uint16_t* wqb   = (uint16_t*)ws;  ws += (size_t)QKV_DIM * EMBED * 2;
  uint16_t* wob   = (uint16_t*)ws;  ws += (size_t)EMBED * EMBED * 2;
  float*    qkv   = (float*)ws;     ws += (size_t)MROWS * QKV_DIM * 4;
  uint16_t* qb    = (uint16_t*)ws;  ws += (size_t)B_ * NHEADS * S_ * HEAD_DIM * 2;
  uint16_t* kb    = (uint16_t*)ws;  ws += (size_t)B_ * NKV * S_ * HEAD_DIM * 2;
  uint16_t* vb    = (uint16_t*)ws;  ws += (size_t)B_ * NKV * S_ * HEAD_DIM * 2;
  uint16_t* attnb = (uint16_t*)ws;  ws += (size_t)MROWS * EMBED * 2;

  int n;
  n = MROWS * EMBED;   cvt_bf16_kernel<<<(n + 255) / 256, 256, 0, stream>>>(x, xb, n);
  n = QKV_DIM * EMBED; cvt_bf16_kernel<<<(n + 255) / 256, 256, 0, stream>>>(w_qkv, wqb, n);
  n = EMBED * EMBED;   cvt_bf16_kernel<<<(n + 255) / 256, 256, 0, stream>>>(w_out, wob, n);

  {  // QKV projection: [4096 x 4096] x [6144 x 4096]^T
    int waves = (MROWS / 32) * (QKV_DIM / 64);
    gemm_bt_wmma<<<waves / 8, 256, 0, stream>>>(xb, wqb, qkv, MROWS, QKV_DIM, EMBED);
  }
  {  // RoPE + scatter to head-major bf16 Q/K (V transposed)
    int total = B_ * S_ * NKV * 6 * 64;
    rope_scatter_kernel<<<total / 256, 256, 0, stream>>>(qkv, qb, kb, vb);
  }
  {  // causal flash attention
    int waves = B_ * NHEADS * (S_ / 16);
    flash_attn_kernel<<<waves / 8, 256, 0, stream>>>(qb, kb, vb, attnb);
  }
  {  // output projection: [4096 x 4096] x [4096 x 4096]^T -> fp32 d_out
    int waves = (MROWS / 32) * (EMBED / 64);
    gemm_bt_wmma<<<waves / 8, 256, 0, stream>>>(attnb, wob, out, MROWS, EMBED, EMBED);
  }
}
